// SelfAttention_9277129359921
// MI455X (gfx1250) — compile-verified, matrix operand order checked by reading, not run on hardware
//
#include <hip/hip_runtime.h>

// ---------------------------------------------------------------------------
// Causal multi-head self-attention for MI455X (gfx1250, wave32, WMMA + TDM).
// B=2, S=4096, D=768, H=12, HD=64.  bf16 WMMA 16x16x32, f32 accumulate.
//   1) cast x, Wq, Wk, Wv, Wo  f32 -> bf16 (RNE)
//   2) QKV projection (WMMA, software-pipelined k-loop)
//   3) flash attention: K/V^T tiles staged to LDS by the Tensor Data Mover
//      (tensor_load_to_lds + s_wait_tensorcnt), double-buffered; online softmax
//   4) output projection (WMMA, pipelined) -> fp32 + bias
// ---------------------------------------------------------------------------

#define BSZ 2
#define SEQ 4096
#define DM  768
#define NH  12
#define HDM 64
#define LOG2E 1.4426950408889634f

typedef unsigned short u16_t;
typedef unsigned int   u32_t;
typedef __attribute__((ext_vector_type(16))) __bf16 v16bf;
typedef __attribute__((ext_vector_type(8)))  float  v8f;
typedef __attribute__((ext_vector_type(4)))  u32_t  v4u;
typedef __attribute__((ext_vector_type(4)))  u32_t  tdm_v4u;
typedef __attribute__((ext_vector_type(8)))  int    tdm_v8i;
typedef __attribute__((ext_vector_type(4)))  int    tdm_v4i;

union FragU { v16bf bf; v4u u[2]; };

__device__ __forceinline__ u16_t f2bf(float f) {
  u32_t u = __builtin_bit_cast(u32_t, f);
  u = (u + 0x7FFFu + ((u >> 16) & 1u)) >> 16;   // round-to-nearest-even
  return (u16_t)u;
}

// 16x32 bf16 fragment (A layout per ISA 7.12.2; B uses same mapping with row=N).
__device__ __forceinline__ v16bf load_frag(const u16_t* base, int stride, int lane) {
  const int row = lane & 15;
  const int kh  = (lane >> 4) & 1;
  const u16_t* p = base + row * stride + kh * 8;
  FragU f;
  f.u[0] = *(const v4u*)(p);
  f.u[1] = *(const v4u*)(p + 16);
  return f.bf;
}

__device__ __forceinline__ v8f wmma_bf16(v16bf a, v16bf b, v8f c) {
  return __builtin_amdgcn_wmma_f32_16x16x32_bf16(false, a, false, b,
                                                 (short)0, c, false, false);
}

// ---------------------------------------------------------------------------
// Tensor Data Mover: 2D bf16 tile Global -> LDS (D# per ISA 08_async_tensor §8)
// ---------------------------------------------------------------------------
#if __has_builtin(__builtin_amdgcn_tensor_load_to_lds)
#define USE_TDM 1
#else
#define USE_TDM 0
#endif

#if USE_TDM
__device__ __forceinline__ void tdm_load_2d(u32_t lds_byte_addr, const void* gptr,
                                            u32_t tile_d0, u32_t tile_d1,
                                            u32_t tensor_d0, u32_t tensor_d1,
                                            unsigned long long stride0_elts) {
  const unsigned long long ga = (unsigned long long)gptr;
  tdm_v4u g0;
  g0.x = 1u;                                          // count=1 (valid), user mode
  g0.y = lds_byte_addr;                               // lds_addr [63:32]
  g0.z = (u32_t)ga;                                   // global_addr lo
  g0.w = (u32_t)((ga >> 32) & 0x1FFFFFFu) | (2u << 30); // global_addr hi | type=2
  tdm_v8i g1;
  g1[0] = (int)(1u << 16);                            // data_size=1 (2 bytes)
  g1[1] = (int)((tensor_d0 & 0xFFFFu) << 16);         // tensor_dim0[15:0]
  g1[2] = (int)(((tensor_d0 >> 16) & 0xFFFFu) | ((tensor_d1 & 0xFFFFu) << 16));
  g1[3] = (int)(((tensor_d1 >> 16) & 0xFFFFu) | ((tile_d0 & 0xFFFFu) << 16));
  g1[4] = (int)(tile_d1 & 0xFFFFu);                   // tile_dim1 | tile_dim2=0
  g1[5] = (int)(u32_t)stride0_elts;                   // tensor_dim0_stride lo
  g1[6] = (int)((u32_t)(stride0_elts >> 32) & 0xFFFFu); // stride0 hi | stride1=0
  g1[7] = 0;
  const tdm_v4i z4 = {0, 0, 0, 0};
#if __clang_major__ >= 23
  const tdm_v8i z8 = {0, 0, 0, 0, 0, 0, 0, 0};
  __builtin_amdgcn_tensor_load_to_lds(g0, g1, z4, z4, z8, 0);
#else
  __builtin_amdgcn_tensor_load_to_lds(g0, g1, z4, z4, 0);
#endif
}
#endif

__device__ __forceinline__ u32_t lds_addr_of(const void* p) {
  return (u32_t)(unsigned long long)p;   // generic LDS ptr: addr[31:0] = LDS byte offset
}

// ---------------------------------------------------------------------------
// 1) fp32 -> bf16 cast of x and the 4 weight matrices
// ---------------------------------------------------------------------------
__global__ __launch_bounds__(256)
void cast_kernel(const float* __restrict__ x,
                 const float* __restrict__ wq, const float* __restrict__ wk,
                 const float* __restrict__ wv, const float* __restrict__ wo,
                 u16_t* __restrict__ xb,
                 u16_t* __restrict__ wqb, u16_t* __restrict__ wkb,
                 u16_t* __restrict__ wvb, u16_t* __restrict__ wob) {
  const long long NX = (long long)BSZ * SEQ * DM;
  const long long NW = (long long)DM * DM;
  const long long total = NX + 4 * NW;
  for (long long i = blockIdx.x * blockDim.x + threadIdx.x; i < total;
       i += (long long)gridDim.x * blockDim.x) {
    if (i < NX)              { xb [i]             = f2bf(x [i]); }
    else if (i < NX + NW)    { wqb[i - NX]        = f2bf(wq[i - NX]); }
    else if (i < NX + 2*NW)  { wkb[i - NX - NW]   = f2bf(wk[i - NX - NW]); }
    else if (i < NX + 3*NW)  { wvb[i - NX - 2*NW] = f2bf(wv[i - NX - 2*NW]); }
    else                     { wob[i - NX - 3*NW] = f2bf(wo[i - NX - 3*NW]); }
  }
}

// ---------------------------------------------------------------------------
// 2) QKV projection, software-pipelined k-loop (double-buffered fragments)
// ---------------------------------------------------------------------------
__global__ __launch_bounds__(256)
void qkv_kernel(const u16_t* __restrict__ xb,
                const u16_t* __restrict__ wqb, const u16_t* __restrict__ wkb,
                const u16_t* __restrict__ wvb,
                const float* __restrict__ bq, const float* __restrict__ bk,
                const float* __restrict__ bv,
                u16_t* __restrict__ Q, u16_t* __restrict__ K,
                u16_t* __restrict__ VT) {
  const int lane   = threadIdx.x & 31;
  const int waveId = blockIdx.x * (blockDim.x >> 5) + (threadIdx.x >> 5);
  const int total  = BSZ * (SEQ / 16) * NH * 3;
  if (waveId >= total) return;

  const int qkv = waveId % 3;
  const int h   = (waveId / 3) % NH;
  const int st  = (waveId / (3 * NH)) % (SEQ / 16);
  const int b   = waveId / (3 * NH * (SEQ / 16));

  const u16_t* W    = (qkv == 0) ? wqb : (qkv == 1) ? wkb : wvb;
  const float* bias = (qkv == 0) ? bq  : (qkv == 1) ? bk  : bv;
  const u16_t* Arow = xb + ((size_t)b * SEQ + (size_t)st * 16) * DM;

  v8f acc[4] = {{}, {}, {}, {}};
  v16bf a_c = load_frag(Arow, DM, lane);
  v16bf b_c[4], b_n[4];
#pragma unroll
  for (int j = 0; j < 4; ++j)
    b_c[j] = load_frag(W + (size_t)(h * HDM + 16 * j) * DM, DM, lane);

#pragma unroll 2
  for (int k = 0; k < DM; k += 32) {
    const int kn = (k + 32) % DM;            // last iteration reloads k=0 (unused)
    const v16bf a_n = load_frag(Arow + kn, DM, lane);
#pragma unroll
    for (int j = 0; j < 4; ++j)
      b_n[j] = load_frag(W + (size_t)(h * HDM + 16 * j) * DM + kn, DM, lane);
#pragma unroll
    for (int j = 0; j < 4; ++j)
      acc[j] = wmma_bf16(a_c, b_c[j], acc[j]);
    a_c = a_n;
#pragma unroll
    for (int j = 0; j < 4; ++j) b_c[j] = b_n[j];
  }

  const int colN = lane & 15;
#pragma unroll
  for (int j = 0; j < 4; ++j) {
#pragma unroll
    for (int r = 0; r < 8; ++r) {
      const int row = (lane < 16) ? r : r + 8;
      const int e   = h * HDM + 16 * j + colN;
      const float v = acc[j][r] + bias[e];
      if (qkv == 2) {
        VT[((size_t)(b * NH + h) * HDM + 16 * j + colN) * SEQ + st * 16 + row] = f2bf(v);
      } else {
        u16_t* dst = (qkv == 0) ? Q : K;
        dst[((size_t)(b * NH + h) * SEQ + st * 16 + row) * HDM + 16 * j + colN] = f2bf(v);
      }
    }
  }
}

// ---------------------------------------------------------------------------
// 3) flash attention: TDM stages K (32x64) and V^T (64x32) tiles into LDS,
//    double-buffered; WMMA for QK^T and P*V; online softmax in registers.
// ---------------------------------------------------------------------------
#define ATTN_WAVES 4
__global__ __launch_bounds__(ATTN_WAVES * 32)
void attn_kernel(const u16_t* __restrict__ Q, const u16_t* __restrict__ K,
                 const u16_t* __restrict__ VT, u16_t* __restrict__ Y) {
  __shared__ __align__(16) u16_t kbuf[ATTN_WAVES][2][32 * HDM];  // 4 KB/tile
  __shared__ __align__(16) u16_t vbuf[ATTN_WAVES][2][HDM * 32];  // 4 KB/tile
  __shared__ __align__(16) u16_t pbuf[ATTN_WAVES][16 * 32];      // 1 KB/wave

  const int lane   = threadIdx.x & 31;
  const int w      = threadIdx.x >> 5;
  const int waveId = blockIdx.x * (blockDim.x >> 5) + w;
  const int total  = BSZ * NH * (SEQ / 16);
  if (waveId >= total) return;

  const int qt = waveId % (SEQ / 16);
  const int h  = (waveId / (SEQ / 16)) % NH;
  const int b  = waveId / ((SEQ / 16) * NH);
  const int q0 = qt * 16;

  const u16_t* Qh  = Q  + (size_t)(b * NH + h) * SEQ * HDM;
  const u16_t* Kh  = K  + (size_t)(b * NH + h) * SEQ * HDM;
  const u16_t* VTh = VT + (size_t)(b * NH + h) * HDM * SEQ;

  v16bf qf[2];
  qf[0] = load_frag(Qh + (size_t)q0 * HDM + 0,  HDM, lane);
  qf[1] = load_frag(Qh + (size_t)q0 * HDM + 32, HDM, lane);

  v8f accO[4] = {{}, {}, {}, {}};
  float m_c[8], l_c[8];
#pragma unroll
  for (int r = 0; r < 8; ++r) { m_c[r] = -INFINITY; l_c[r] = 0.0f; }

  u16_t* myLds = &pbuf[w][0];
  const int colN  = lane & 15;
  const int rowHi = (lane < 16) ? 0 : 8;
  const float sc  = 0.125f;                 // 1/sqrt(64)

  const int nChunks = (q0 + 16 + 31) / 32;

#if USE_TDM
  // prologue: stage chunk 0 into buffer 0
  tdm_load_2d(lds_addr_of(&kbuf[w][0][0]), Kh, HDM, 32, HDM, SEQ, HDM);
  tdm_load_2d(lds_addr_of(&vbuf[w][0][0]), VTh, 32, HDM, SEQ, HDM, SEQ);
#endif

  int buf = 0;
  for (int c = 0; c < nChunks; ++c) {
    const int kb = c * 32;
    const bool more = (c + 1 < nChunks);

#if USE_TDM
    if (more) {
      const int kb2 = kb + 32;
      tdm_load_2d(lds_addr_of(&kbuf[w][buf ^ 1][0]),
                  Kh + (size_t)kb2 * HDM, HDM, 32, HDM, SEQ, HDM);
      tdm_load_2d(lds_addr_of(&vbuf[w][buf ^ 1][0]),
                  VTh + kb2, 32, HDM, SEQ, HDM, SEQ);
      __builtin_amdgcn_s_wait_tensorcnt(2);   // current buffer's 2 loads done
    } else {
      __builtin_amdgcn_s_wait_tensorcnt(0);
    }
    const u16_t* kTile = &kbuf[w][buf][0];    // 32 rows x 64, stride 64
    const u16_t* vTile = &vbuf[w][buf][0];    // 64 rows x 32, stride 32
    const int kStride = HDM, vStride = 32, kOff = 0;
#else
    const u16_t* kTile = Kh + (size_t)kb * HDM;
    const u16_t* vTile = VTh;
    const int kStride = HDM, vStride = SEQ, kOff = kb;
#endif

    // ---- S = (Q K^T) * sc with causal mask ----
    v8f accS[2] = {{}, {}};
#pragma unroll
    for (int nt = 0; nt < 2; ++nt) {
#pragma unroll
      for (int d = 0; d < 2; ++d) {
        const v16bf kf = load_frag(kTile + (size_t)(16 * nt) * kStride + 32 * d,
                                   kStride, lane);
        accS[nt] = wmma_bf16(qf[d], kf, accS[nt]);
      }
    }

    float sreg[2][8];
#pragma unroll
    for (int nt = 0; nt < 2; ++nt)
#pragma unroll
      for (int r = 0; r < 8; ++r) {
        const int row = q0 + rowHi + r;
        const int col = kb + 16 * nt + colN;
        const float s = accS[nt][r] * sc;
        sreg[nt][r] = (col > row) ? -INFINITY : s;
      }

    // ---- online softmax (row reductions across the 16-lane half-group) ----
    float p[2][8];
#pragma unroll
    for (int r = 0; r < 8; ++r) {
      float v = fmaxf(sreg[0][r], sreg[1][r]);
      v = fmaxf(v, __shfl_xor(v, 1, 32));
      v = fmaxf(v, __shfl_xor(v, 2, 32));
      v = fmaxf(v, __shfl_xor(v, 4, 32));
      v = fmaxf(v, __shfl_xor(v, 8, 32));
      const float m_new = fmaxf(m_c[r], v);
      const float alpha = exp2f((m_c[r] - m_new) * LOG2E);
      m_c[r] = m_new;
      p[0][r] = exp2f((sreg[0][r] - m_new) * LOG2E);
      p[1][r] = exp2f((sreg[1][r] - m_new) * LOG2E);
      float rs = p[0][r] + p[1][r];
      rs += __shfl_xor(rs, 1, 32);
      rs += __shfl_xor(rs, 2, 32);
      rs += __shfl_xor(rs, 4, 32);
      rs += __shfl_xor(rs, 8, 32);
      l_c[r] = l_c[r] * alpha + rs;
#pragma unroll
      for (int j = 0; j < 4; ++j) accO[j][r] *= alpha;
    }

    // ---- P (C-layout) -> LDS 16x32 bf16 -> A-layout fragment ----
#pragma unroll
    for (int nt = 0; nt < 2; ++nt)
#pragma unroll
      for (int r = 0; r < 8; ++r)
        myLds[(rowHi + r) * 32 + 16 * nt + colN] = f2bf(p[nt][r]);
    asm volatile("s_wait_dscnt 0" ::: "memory");
    const v16bf pf = load_frag(myLds, 32, lane);

    // ---- O += P @ V ----
#pragma unroll
    for (int j = 0; j < 4; ++j) {
      const v16bf vf = load_frag(vTile + (size_t)(16 * j) * vStride + kOff,
                                 vStride, lane);
      accO[j] = wmma_bf16(pf, vf, accO[j]);
    }
    asm volatile("s_wait_dscnt 0" ::: "memory");
    buf ^= 1;
  }

  // ---- normalize and store y[b, s, h*64 + d] ----
  float inv_l[8];
#pragma unroll
  for (int r = 0; r < 8; ++r) inv_l[r] = 1.0f / l_c[r];
#pragma unroll
  for (int j = 0; j < 4; ++j)
#pragma unroll
    for (int r = 0; r < 8; ++r) {
      const int row = rowHi + r;
      const float o = accO[j][r] * inv_l[r];
      Y[((size_t)b * SEQ + q0 + row) * DM + h * HDM + 16 * j + colN] = f2bf(o);
    }
}

// ---------------------------------------------------------------------------
// 4) output projection: out = y @ Wo^T + bo (fp32), pipelined k-loop
// ---------------------------------------------------------------------------
__global__ __launch_bounds__(256)
void oproj_kernel(const u16_t* __restrict__ Y, const u16_t* __restrict__ wob,
                  const float* __restrict__ bo, float* __restrict__ out) {
  const int lane   = threadIdx.x & 31;
  const int waveId = blockIdx.x * (blockDim.x >> 5) + (threadIdx.x >> 5);
  const int total  = BSZ * (SEQ / 16) * (DM / 64);
  if (waveId >= total) return;

  const int o64 = waveId % (DM / 64);
  const int st  = (waveId / (DM / 64)) % (SEQ / 16);
  const int b   = waveId / ((DM / 64) * (SEQ / 16));

  const u16_t* Arow = Y + ((size_t)b * SEQ + (size_t)st * 16) * DM;
  const u16_t* Wb   = wob + (size_t)(o64 * 64) * DM;

  v8f acc[4] = {{}, {}, {}, {}};
  v16bf a_c = load_frag(Arow, DM, lane);
  v16bf b_c[4], b_n[4];
#pragma unroll
  for (int j = 0; j < 4; ++j)
    b_c[j] = load_frag(Wb + (size_t)(16 * j) * DM, DM, lane);

#pragma unroll 2
  for (int k = 0; k < DM; k += 32) {
    const int kn = (k + 32) % DM;
    const v16bf a_n = load_frag(Arow + kn, DM, lane);
#pragma unroll
    for (int j = 0; j < 4; ++j)
      b_n[j] = load_frag(Wb + (size_t)(16 * j) * DM + kn, DM, lane);
#pragma unroll
    for (int j = 0; j < 4; ++j)
      acc[j] = wmma_bf16(a_c, b_c[j], acc[j]);
    a_c = a_n;
#pragma unroll
    for (int j = 0; j < 4; ++j) b_c[j] = b_n[j];
  }

  const int colN = lane & 15;
#pragma unroll
  for (int j = 0; j < 4; ++j)
#pragma unroll
    for (int r = 0; r < 8; ++r) {
      const int row = (lane < 16) ? r : r + 8;
      const int e   = o64 * 64 + 16 * j + colN;
      out[((size_t)b * SEQ + st * 16 + row) * DM + e] = acc[j][r] + bo[e];
    }
}

// ---------------------------------------------------------------------------
// launch
// ---------------------------------------------------------------------------
extern "C" void kernel_launch(void* const* d_in, const int* in_sizes, int n_in,
                              void* d_out, int out_size, void* d_ws, size_t ws_size,
                              hipStream_t stream) {
  const float* x  = (const float*)d_in[0];
  const float* Wq = (const float*)d_in[1];
  const float* bq = (const float*)d_in[2];
  const float* Wk = (const float*)d_in[3];
  const float* bk = (const float*)d_in[4];
  const float* Wv = (const float*)d_in[5];
  const float* bv = (const float*)d_in[6];
  const float* Wo = (const float*)d_in[7];
  const float* bo = (const float*)d_in[8];
  float* out = (float*)d_out;

  const size_t NX = (size_t)BSZ * SEQ * DM;   // elements
  const size_t NW = (size_t)DM * DM;

  u16_t* ws  = (u16_t*)d_ws;                  // bf16 workspace regions
  u16_t* xb  = ws;            ws += NX;
  u16_t* wqb = ws;            ws += NW;
  u16_t* wkb = ws;            ws += NW;
  u16_t* wvb = ws;            ws += NW;
  u16_t* wob = ws;            ws += NW;
  u16_t* Qb  = ws;            ws += NX;
  u16_t* Kb  = ws;            ws += NX;
  u16_t* VTb = ws;            ws += NX;
  u16_t* Yb  = ws;            ws += NX;
  (void)ws_size; (void)in_sizes; (void)n_in; (void)out_size;

  cast_kernel<<<2048, 256, 0, stream>>>(x, Wq, Wk, Wv, Wo, xb, wqb, wkb, wvb, wob);

  {
    const int waves = BSZ * (SEQ / 16) * NH * 3;           // 18432
    qkv_kernel<<<(waves + 7) / 8, 256, 0, stream>>>(xb, wqb, wkb, wvb, bq, bk, bv,
                                                    Qb, Kb, VTb);
  }
  {
    const int waves = BSZ * NH * (SEQ / 16);               // 6144
    attn_kernel<<<(waves + ATTN_WAVES - 1) / ATTN_WAVES, ATTN_WAVES * 32, 0,
                  stream>>>(Qb, Kb, VTb, Yb);
  }
  {
    const int waves = BSZ * (SEQ / 16) * (DM / 64);        // 6144
    oproj_kernel<<<(waves + 7) / 8, 256, 0, stream>>>(Yb, wob, bo, out);
  }
}
